// SelfAttention_5222680232307
// MI455X (gfx1250) — matrix operand order repaired and checked by reading, not compiled, not run on hardware
//
#include <hip/hip_runtime.h>
#include <hip/hip_bf16.h>
#include <math.h>
#include <cstdint>

// Problem constants (from reference): x[B,S,H], W[H,H], out[B,S,H]
static constexpr int HD = 1024;       // hidden
static constexpr int SQ = 4096;       // sequence
static constexpr int BT = 4;          // batch
static constexpr int MT = BT * SQ;    // total rows = 16384

typedef __attribute__((ext_vector_type(16))) __bf16          v16bf;
typedef __attribute__((ext_vector_type(16))) unsigned short  v16u;
typedef __attribute__((ext_vector_type(8)))  unsigned short  v8u;
typedef __attribute__((ext_vector_type(8)))  float           v8f;
typedef __attribute__((ext_vector_type(4)))  unsigned int    v4ui;
typedef __attribute__((ext_vector_type(8)))  int             v8i;
typedef __attribute__((ext_vector_type(4)))  int             v4i;

// Native fp32 -> bf16 convert (single VALU op on gfx1250).
__device__ __forceinline__ unsigned short f2bf(float f) {
  return __builtin_bit_cast(unsigned short, (__bf16)f);
}

__device__ __forceinline__ v8f zero8() {
  v8f z;
  #pragma unroll
  for (int i = 0; i < 8; ++i) z[i] = 0.0f;
  return z;
}

__device__ __forceinline__ v8f wmma_bf16(v16u a, v16u b, v8f c) {
  return __builtin_amdgcn_wmma_f32_16x16x32_bf16(
      false, __builtin_bit_cast(v16bf, a),
      false, __builtin_bit_cast(v16bf, b),
      (short)0, c, false, false);
}

// Join two aligned 16-byte halves into one 32-byte WMMA operand.
__device__ __forceinline__ v16u ldfrag(const unsigned short* p0,
                                       const unsigned short* p1) {
  v8u lo = *(const v8u*)p0;
  v8u hi = *(const v8u*)p1;
  return __builtin_shufflevector(lo, hi, 0, 1, 2, 3, 4, 5, 6, 7,
                                 8, 9, 10, 11, 12, 13, 14, 15);
}

// ---------------------------------------------------------------------------
// Tensor Data Mover: 2-D tile load (bf16 elements) global -> LDS.
// D# layout per CDNA5 ISA §8 (group0: count/lds/global/type, group1:
// data_size, tensor_dim0/1, tile_dim0/1, dim0 stride). Groups 2/3 zero (2-D).
// ---------------------------------------------------------------------------
#if __has_builtin(__builtin_amdgcn_tensor_load_to_lds)
#define HAVE_TDM 1
__device__ __forceinline__ void tdm_load_2d_bf16(unsigned lds_addr, const void* gaddr,
                                                 unsigned tensor_d0, unsigned tensor_d1,
                                                 unsigned tile_d0, unsigned tile_d1,
                                                 unsigned long long stride0_elts) {
  unsigned long long ga = (unsigned long long)(uintptr_t)gaddr;
  v4ui g0;
  g0[0] = 1u;                                             // count=1 valid user D#
  g0[1] = lds_addr;                                       // LDS byte address
  g0[2] = (unsigned)(ga & 0xFFFFFFFFu);                   // global_addr[31:0]
  g0[3] = (unsigned)((ga >> 32) & 0x1FFFFFFu) | (2u << 30); // ga[56:32] | type=2
  v8i g1;
  g1[0] = (int)(1u << 16);                                // data_size=1 (2 bytes)
  g1[1] = (int)((tensor_d0 & 0xFFFFu) << 16);             // tensor_dim0 lo16
  g1[2] = (int)((tensor_d0 >> 16) | ((tensor_d1 & 0xFFFFu) << 16));
  g1[3] = (int)((tensor_d1 >> 16) | (tile_d0 << 16));     // tile_dim0
  g1[4] = (int)(tile_d1 & 0xFFFFu);                       // tile_dim1 (tile_dim2=0)
  g1[5] = (int)(unsigned)(stride0_elts & 0xFFFFFFFFu);    // dim0 stride lo32
  g1[6] = (int)(unsigned)((stride0_elts >> 32) & 0xFFFFu);// dim0 stride hi16
  g1[7] = 0;
  v4i g2 = {0, 0, 0, 0};
  v4i g3 = {0, 0, 0, 0};
#if defined(__clang_major__) && (__clang_major__ >= 23)
  v8i g4 = {0, 0, 0, 0, 0, 0, 0, 0};
  __builtin_amdgcn_tensor_load_to_lds(g0, g1, g2, g3, g4, 0);
#else
  __builtin_amdgcn_tensor_load_to_lds(g0, g1, g2, g3, 0);
#endif
}
#else
#define HAVE_TDM 0
#endif

// ---------------------------------------------------------------------------
// Kernel 1: y = x @ W^T + b  (fp32 in, bf16 out), one launch per Q/K/V.
// Block = 256 threads (8 waves). Block tile 128(M) x 64(N), wave tile 32x32.
// outT (optional) additionally stores y transposed per-batch [HD][SQ].
// ---------------------------------------------------------------------------
__global__ __launch_bounds__(256)
void qkv_gemm_kernel(const float* __restrict__ x, const float* __restrict__ W,
                     const float* __restrict__ bias, unsigned short* __restrict__ out,
                     unsigned short* __restrict__ outT) {
  __shared__ unsigned short As[128][40];   // x tile, 80B row stride (16B-aligned runs)
  __shared__ unsigned short Bt[32][80];    // transposed W tile: Bt[k][n], 160B stride

  const int tid  = threadIdx.x;
  const int lane = tid & 31;
  const int wave = tid >> 5;
  const int m0 = blockIdx.x * 128;
  const int n0 = blockIdx.y * 64;
  const int wm = wave >> 1;   // 0..3
  const int wn = wave & 1;    // 0..1

  v8f acc[2][2];
  #pragma unroll
  for (int i = 0; i < 2; ++i)
    #pragma unroll
    for (int j = 0; j < 2; ++j) acc[i][j] = zero8();

  const int arow = lane & 15;
  const int khi  = (lane >> 4) << 3;   // 0 or 8

  // fixed per-thread staging slices (no loops, no per-element addressing)
  const int sr  = tid >> 1;            // As row 0..127
  const int sc  = (tid & 1) * 16;      // As col 0 or 16
  const int wn_r = tid >> 2;           // W row (N) 0..63
  const int wc   = (tid & 3) * 8;      // W col chunk 0/8/16/24

  #pragma unroll 1
  for (int k0 = 0; k0 < HD; k0 += 32) {
    // stage x tile 128x32: 4x float4 loads -> 2x 16B LDS stores per thread
    {
      const float4* src = (const float4*)(x + (size_t)(m0 + sr) * HD + k0 + sc);
      float4 f0 = src[0], f1 = src[1], f2c = src[2], f3 = src[3];
      v8u lo, hi;
      lo[0] = f2bf(f0.x); lo[1] = f2bf(f0.y); lo[2] = f2bf(f0.z); lo[3] = f2bf(f0.w);
      lo[4] = f2bf(f1.x); lo[5] = f2bf(f1.y); lo[6] = f2bf(f1.z); lo[7] = f2bf(f1.w);
      hi[0] = f2bf(f2c.x); hi[1] = f2bf(f2c.y); hi[2] = f2bf(f2c.z); hi[3] = f2bf(f2c.w);
      hi[4] = f2bf(f3.x); hi[5] = f2bf(f3.y); hi[6] = f2bf(f3.z); hi[7] = f2bf(f3.w);
      *(v8u*)&As[sr][sc]     = lo;
      *(v8u*)&As[sr][sc + 8] = hi;
    }
    // stage W tile 64x32 transposed: Bt[k][n] = W[n0+n][k0+k]
    {
      const float4* src = (const float4*)(W + (size_t)(n0 + wn_r) * HD + k0 + wc);
      float4 f0 = src[0], f1 = src[1];
      Bt[wc + 0][wn_r] = f2bf(f0.x); Bt[wc + 1][wn_r] = f2bf(f0.y);
      Bt[wc + 2][wn_r] = f2bf(f0.z); Bt[wc + 3][wn_r] = f2bf(f0.w);
      Bt[wc + 4][wn_r] = f2bf(f1.x); Bt[wc + 5][wn_r] = f2bf(f1.y);
      Bt[wc + 6][wn_r] = f2bf(f1.z); Bt[wc + 7][wn_r] = f2bf(f1.w);
    }
    if (k0 + 32 < HD) {
      // speculative prefetch of next x chunk -> global_prefetch_b8
      __builtin_prefetch(&x[(size_t)(m0 + sr) * HD + k0 + 32 + sc], 0, 3);
    }
    __syncthreads();

    // A fragments: lane<16 holds K {0..7,16..23}, lane>=16 holds {8..15,24..31}
    v16u afr[2];
    #pragma unroll
    for (int i = 0; i < 2; ++i) {
      int rr = wm * 32 + i * 16 + arow;
      afr[i] = ldfrag(&As[rr][khi], &As[rr][16 + khi]);
    }
    // B fragments: lane&15 = N; elements gathered over K in table order
    v16u bfr[2];
    #pragma unroll
    for (int j = 0; j < 2; ++j)
      #pragma unroll
      for (int e = 0; e < 16; ++e)
        bfr[j][e] = Bt[(e < 8 ? khi + e : 8 + khi + e)][wn * 32 + j * 16 + arow];

    #pragma unroll
    for (int i = 0; i < 2; ++i)
      #pragma unroll
      for (int j = 0; j < 2; ++j)
        acc[i][j] = wmma_bf16(afr[i], bfr[j], acc[i][j]);
    __syncthreads();
  }

  // epilogue: C/D layout -> VGPR r, lane<16: M=r ; lane>=16: M=r+8 ; N=lane&15
  const int rl = lane & 15;
  const int rh = (lane >> 4) << 3;
  #pragma unroll
  for (int i = 0; i < 2; ++i) {
    #pragma unroll
    for (int j = 0; j < 2; ++j) {
      int col = n0 + wn * 32 + j * 16 + rl;
      float bv = bias[col];
      #pragma unroll
      for (int r = 0; r < 8; ++r) {
        int row = m0 + wm * 32 + i * 16 + r + rh;
        unsigned short val = f2bf(acc[i][j][r] + bv);
        out[(size_t)row * HD + col] = val;
        if (outT) {  // per-batch transposed copy: [b][col][s]
          int bb = row >> 12;           // row / SQ
          int s  = row & (SQ - 1);
          outT[((size_t)bb * HD + col) * SQ + s] = val;
        }
      }
    }
  }
}

// ---------------------------------------------------------------------------
// Kernel 2: streaming softmax row statistics (max, sum-exp) of Q K^T.
// Q tile via TDM; transposed K (Kt[b][h][s]) gives contiguous 32B B-fragments.
// ---------------------------------------------------------------------------
__global__ __launch_bounds__(256)
void softmax_stats_kernel(const unsigned short* __restrict__ Qb,
                          const unsigned short* __restrict__ Kt,
                          float* __restrict__ mrow, float* __restrict__ lrow) {
  __shared__ unsigned short Qs[16][HD];    // 32 KB, 2048B row stride
  __shared__ float red[8][16][2];

  const int tid  = threadIdx.x;
  const int lane = tid & 31;
  const int wave = tid >> 5;
  const int q0 = blockIdx.x * 16;
  const int b  = blockIdx.y;
  const size_t mbase = (size_t)b * SQ + q0;

#if HAVE_TDM
  if (wave == 0) {
    tdm_load_2d_bf16((unsigned)(uintptr_t)&Qs[0][0], Qb + mbase * HD,
                     HD, 16, HD, 16, (unsigned long long)HD);
    __builtin_amdgcn_s_wait_tensorcnt(0);
  }
#else
  for (int i = tid; i < 16 * HD; i += 256)
    Qs[i >> 10][i & (HD - 1)] = Qb[(mbase + (i >> 10)) * HD + (i & (HD - 1))];
#endif
  __syncthreads();

  float mloc[8], lloc[8];
  #pragma unroll
  for (int r = 0; r < 8; ++r) { mloc[r] = -1e30f; lloc[r] = 0.0f; }

  const int arow = lane & 15;
  const int khi  = (lane >> 4) << 3;

  #pragma unroll 1
  for (int t = wave; t < SQ / 16; t += 8) {
    const int kb = t * 16;
    v8f s = zero8();
    #pragma unroll 1
    for (int h0 = 0; h0 < HD; h0 += 32) {
      v16u au = ldfrag(&Qs[arow][h0 + khi], &Qs[arow][h0 + 16 + khi]);
      // B[k][n] = K[kb+n][h0+k] = Kt[b][h0+k][kb+n]: lane&15 = n, elements over k (table order)
      v16u bu;
      #pragma unroll
      for (int e = 0; e < 16; ++e)
        bu[e] = Kt[((size_t)b * HD + h0 + (e < 8 ? khi + e : 8 + khi + e)) * SQ + kb + arow];
      s = wmma_bf16(au, bu, s);
    }
    #pragma unroll
    for (int r = 0; r < 8; ++r) {
      float v  = s[r];
      float nm = fmaxf(mloc[r], v);
      lloc[r]  = lloc[r] * __expf(mloc[r] - nm) + __expf(v - nm);
      mloc[r]  = nm;
    }
  }

  // reduce across the 16 lanes of each half-group (cols of the score tile)
  #pragma unroll
  for (int r = 0; r < 8; ++r) {
    float m = mloc[r], l = lloc[r];
    #pragma unroll
    for (int off = 8; off >= 1; off >>= 1) {
      float om = __shfl_xor(m, off, 32);
      float ol = __shfl_xor(l, off, 32);
      float nm = fmaxf(m, om);
      l = l * __expf(m - nm) + ol * __expf(om - nm);
      m = nm;
    }
    mloc[r] = m; lloc[r] = l;
  }
  if ((lane & 15) == 0) {
    int rbase = (lane >> 4) << 3;
    #pragma unroll
    for (int r = 0; r < 8; ++r) {
      red[wave][rbase + r][0] = mloc[r];
      red[wave][rbase + r][1] = lloc[r];
    }
  }
  __syncthreads();
  if (tid < 16) {
    float m = -1e30f;
    for (int w = 0; w < 8; ++w) m = fmaxf(m, red[w][tid][0]);
    float l = 0.0f;
    for (int w = 0; w < 8; ++w) l += red[w][tid][1] * __expf(red[w][tid][0] - m);
    mrow[mbase + tid] = m;
    lrow[mbase + tid] = l;
  }
}

// ---------------------------------------------------------------------------
// Kernel 3: out = softmax(QK^T) V. Block = 1024 (32 waves) per 16-row q-tile.
// Q tile is DMA'd into LDS by the Tensor Data Mover; per 512-key chunk each
// wave builds one 16x16 exp-score tile in LDS, then all waves do P@V for
// their own 32 output columns.
// ---------------------------------------------------------------------------
__global__ __launch_bounds__(1024)
void attn_out_kernel(const unsigned short* __restrict__ Qb,
                     const unsigned short* __restrict__ Kt,
                     const unsigned short* __restrict__ Vb,
                     const float* __restrict__ mrow, const float* __restrict__ lrow,
                     float* __restrict__ out) {
  __shared__ unsigned short Qs[16][HD];         // 32 KB
  __shared__ unsigned short Ps[16][520];        // exp-score chunk, 1040B stride

  const int tid  = threadIdx.x;
  const int lane = tid & 31;
  const int wave = tid >> 5;                    // 0..31
  const int q0 = blockIdx.x * 16;
  const int b  = blockIdx.y;
  const size_t mbase = (size_t)b * SQ + q0;

#if HAVE_TDM
  if (wave == 0) {
    // 2-D TDM: 16 rows x 1024 bf16, row stride HD -> contiguous into Qs
    tdm_load_2d_bf16((unsigned)(uintptr_t)&Qs[0][0], Qb + mbase * HD,
                     HD, 16, HD, 16, (unsigned long long)HD);
    __builtin_amdgcn_s_wait_tensorcnt(0);
  }
#else
  for (int i = tid; i < 16 * HD; i += 1024)
    Qs[i >> 10][i & (HD - 1)] = Qb[(mbase + (i >> 10)) * HD + (i & (HD - 1))];
#endif
  __syncthreads();

  const int arow = lane & 15;
  const int khi  = (lane >> 4) << 3;
  const int rh   = (lane >> 4) << 3;

  float mr[8], lr[8];
  #pragma unroll
  for (int r = 0; r < 8; ++r) {
    mr[r] = mrow[mbase + r + rh];
    lr[r] = lrow[mbase + r + rh];
  }

  const int n0 = wave * 32;                     // this wave's 32 output columns
  v8f o0 = zero8(), o1 = zero8();

  #pragma unroll 1
  for (int kc = 0; kc < SQ; kc += 512) {
    // ---- phase 1: score tile for keys [kc + wave*16, +16) ----
    const int kb = kc + wave * 16;
    v8f s = zero8();
    #pragma unroll 1
    for (int h0 = 0; h0 < HD; h0 += 32) {
      v16u au = ldfrag(&Qs[arow][h0 + khi], &Qs[arow][h0 + 16 + khi]);
      v16u bu;
      #pragma unroll
      for (int e = 0; e < 16; ++e)
        bu[e] = Kt[((size_t)b * HD + h0 + (e < 8 ? khi + e : 8 + khi + e)) * SQ + kb + arow];
      s = wmma_bf16(au, bu, s);
    }
    #pragma unroll
    for (int r = 0; r < 8; ++r) {
      float p = __expf(s[r] - mr[r]);
      Ps[r + rh][wave * 16 + (lane & 15)] = f2bf(p);
    }
    __syncthreads();

    // ---- phase 2: P(16x512) @ V(512x32) for this wave's columns ----
    #pragma unroll 1
    for (int kk = 0; kk < 512; kk += 32) {
      v16u au = ldfrag(&Ps[arow][kk + khi], &Ps[arow][kk + 16 + khi]);
      v16u b0, b1;
      #pragma unroll
      for (int e = 0; e < 16; ++e) {
        const size_t vrow = (size_t)b * SQ + kc + kk + (e < 8 ? khi + e : 8 + khi + e);   // element = K row of B (table order)
        b0[e] = Vb[vrow * HD + n0 + arow];                   // lane&15 = N
        b1[e] = Vb[vrow * HD + n0 + 16 + arow];
      }
      o0 = wmma_bf16(au, b0, o0);
      o1 = wmma_bf16(au, b1, o1);
    }
    __syncthreads();
  }

  // epilogue: divide by row sum, store fp32
  #pragma unroll
  for (int r = 0; r < 8; ++r) {
    float inv = 1.0f / lr[r];
    size_t row = mbase + r + rh;
    out[row * HD + n0 + (lane & 15)]      = o0[r] * inv;
    out[row * HD + n0 + 16 + (lane & 15)] = o1[r] * inv;
  }
}

// ---------------------------------------------------------------------------
extern "C" void kernel_launch(void* const* d_in, const int* in_sizes, int n_in,
                              void* d_out, int out_size, void* d_ws, size_t ws_size,
                              hipStream_t stream) {
  (void)in_sizes; (void)n_in; (void)out_size; (void)ws_size;
  const float* x  = (const float*)d_in[0];
  const float* Wq = (const float*)d_in[1];
  const float* bq = (const float*)d_in[2];
  const float* Wk = (const float*)d_in[3];
  const float* bk = (const float*)d_in[4];
  const float* Wv = (const float*)d_in[5];
  const float* bv = (const float*)d_in[6];
  float* out = (float*)d_out;

  // workspace layout: Qb | Kb | Vb | Kt (bf16 bits) | mrow | lrow
  unsigned short* Qb = (unsigned short*)d_ws;
  unsigned short* Kb = Qb + (size_t)MT * HD;
  unsigned short* Vb = Kb + (size_t)MT * HD;
  unsigned short* Kt = Vb + (size_t)MT * HD;     // per-batch [HD][SQ]
  float* mrow = (float*)(Kt + (size_t)MT * HD);
  float* lrow = mrow + MT;

  dim3 g1(MT / 128, HD / 64);
  qkv_gemm_kernel<<<g1, 256, 0, stream>>>(x, Wq, bq, Qb, nullptr);
  qkv_gemm_kernel<<<g1, 256, 0, stream>>>(x, Wk, bk, Kb, Kt);
  qkv_gemm_kernel<<<g1, 256, 0, stream>>>(x, Wv, bv, Vb, nullptr);

  dim3 g2(SQ / 16, BT);
  softmax_stats_kernel<<<g2, 256, 0, stream>>>(Qb, Kt, mrow, lrow);
  attn_out_kernel<<<g2, 1024, 0, stream>>>(Qb, Kt, Vb, mrow, lrow, out);
}
